// MultiHeadModel_27642409517470
// MI455X (gfx1250) — compile-verified, matrix operand order checked by reading, not run on hardware
//
#include <hip/hip_runtime.h>
#include <hip/hip_bf16.h>

typedef __bf16 bf16;
typedef bf16  v16bf __attribute__((ext_vector_type(16)));
typedef float v8f   __attribute__((ext_vector_type(8)));

union Frag16 {           // one WMMA bf16 A/B fragment: 16 bf16 = 32 bytes
    v16bf v;
    uint4 q[2];
};

// ---------------------------------------------------------------- constants
#define CTX     2048
#define DMODEL  1024
#define NHEADS  16
#define HEADD   64
#define BATCH   2
#define SEQ     2048
#define VOCAB   32000

// ---------------------------------------------------------------- fp32 -> bf16 convert, staged
// through LDS with gfx1250 async global->LDS copies. Requires n % blockDim == 0
// (all launches below satisfy this) so EXEC is all-ones around the async op.
__global__ __launch_bounds__(256) void cvt_bf16_async_kernel(const float* __restrict__ x,
                                                             bf16* __restrict__ y, int n) {
    __shared__ __align__(16) float buf[256];
    int i = blockIdx.x * 256 + threadIdx.x;
    unsigned ldsAddr = (unsigned)(uintptr_t)&buf[threadIdx.x];
    const float* gp = x + i;
    asm volatile("global_load_async_to_lds_b32 %0, %1, off"
                 :: "v"(ldsAddr), "v"(gp) : "memory");
    asm volatile("s_wait_asynccnt 0" ::: "memory");
    __syncthreads();
    y[i] = (bf16)buf[threadIdx.x];
}

// ---------------------------------------------------------------- embedding gather + pos add
__global__ __launch_bounds__(256) void embed_kernel(const int* __restrict__ tok,
                                                    const float* __restrict__ emb,
                                                    const float* __restrict__ pos,
                                                    bf16* __restrict__ xb) {
    int i = blockIdx.x * blockDim.x + threadIdx.x;   // over B*T*C
    if (i >= BATCH * SEQ * DMODEL) return;
    int c = i & (DMODEL - 1);
    int t = (i >> 10) & (SEQ - 1);
    int b = i >> 21;
    int id = tok[b * SEQ + t];
    xb[i] = (bf16)(emb[(size_t)id * DMODEL + c] + pos[(size_t)t * DMODEL + c]);
}

// ---------------------------------------------------------------- generic bf16 WMMA GEMM
// Y[M x N] (f32) = X[M x K] (bf16, row major) @ W[N x K]^T (bf16, row major)
// one wave computes a 32x64 output tile (2 M-subtiles x 4 N-subtiles = 8 WMMAs
// per 32-wide k-step, 6 fragment loads). 4 waves / block.
__global__ __launch_bounds__(128) void gemm_bf16_kernel(const bf16* __restrict__ X,
                                                        const bf16* __restrict__ W,
                                                        float* __restrict__ Y,
                                                        int M, int N, int K) {
    int wave = blockIdx.x * (blockDim.x >> 5) + (threadIdx.x >> 5);
    int lane = threadIdx.x & 31;
    int half = lane >> 4;
    int lr   = lane & 15;
    int tN = N >> 6;                 // 64-wide col blocks
    int tM = M >> 5;                 // 32-row blocks
    if (wave >= tM * tN) return;
    int mb = wave / tN;
    int nb = wave % tN;

    const bf16* xr0 = X + (size_t)(mb * 32 + lr) * K + half * 8;
    const bf16* xr1 = xr0 + (size_t)16 * K;
    const bf16* wr0 = W + (size_t)(nb * 64 + lr) * K + half * 8;
    const bf16* wr1 = wr0 + (size_t)16 * K;
    const bf16* wr2 = wr0 + (size_t)32 * K;
    const bf16* wr3 = wr0 + (size_t)48 * K;

    v8f acc[2][4] = {{{}, {}, {}, {}}, {{}, {}, {}, {}}};
    for (int kb = 0; kb < K; kb += 32) {
        // prefetch next A rows (demonstrates gfx1250 global_prefetch path)
        __builtin_prefetch(xr0 + kb + 256, 0, 1);
        __builtin_prefetch(xr1 + kb + 256, 0, 1);

        Frag16 a0, a1, b0, b1, b2, b3;
        a0.q[0] = *(const uint4*)(xr0 + kb);  a0.q[1] = *(const uint4*)(xr0 + kb + 16);
        a1.q[0] = *(const uint4*)(xr1 + kb);  a1.q[1] = *(const uint4*)(xr1 + kb + 16);
        b0.q[0] = *(const uint4*)(wr0 + kb);  b0.q[1] = *(const uint4*)(wr0 + kb + 16);
        b1.q[0] = *(const uint4*)(wr1 + kb);  b1.q[1] = *(const uint4*)(wr1 + kb + 16);
        b2.q[0] = *(const uint4*)(wr2 + kb);  b2.q[1] = *(const uint4*)(wr2 + kb + 16);
        b3.q[0] = *(const uint4*)(wr3 + kb);  b3.q[1] = *(const uint4*)(wr3 + kb + 16);

        acc[0][0] = __builtin_amdgcn_wmma_f32_16x16x32_bf16(false, a0.v, false, b0.v, (short)0, acc[0][0], false, false);
        acc[0][1] = __builtin_amdgcn_wmma_f32_16x16x32_bf16(false, a0.v, false, b1.v, (short)0, acc[0][1], false, false);
        acc[0][2] = __builtin_amdgcn_wmma_f32_16x16x32_bf16(false, a0.v, false, b2.v, (short)0, acc[0][2], false, false);
        acc[0][3] = __builtin_amdgcn_wmma_f32_16x16x32_bf16(false, a0.v, false, b3.v, (short)0, acc[0][3], false, false);
        acc[1][0] = __builtin_amdgcn_wmma_f32_16x16x32_bf16(false, a1.v, false, b0.v, (short)0, acc[1][0], false, false);
        acc[1][1] = __builtin_amdgcn_wmma_f32_16x16x32_bf16(false, a1.v, false, b1.v, (short)0, acc[1][1], false, false);
        acc[1][2] = __builtin_amdgcn_wmma_f32_16x16x32_bf16(false, a1.v, false, b2.v, (short)0, acc[1][2], false, false);
        acc[1][3] = __builtin_amdgcn_wmma_f32_16x16x32_bf16(false, a1.v, false, b3.v, (short)0, acc[1][3], false, false);
    }

#pragma unroll
    for (int i = 0; i < 2; ++i)
#pragma unroll
        for (int t = 0; t < 4; ++t) {
            float* yr = Y + (size_t)(mb * 32 + i * 16 + half * 8) * N + nb * 64 + t * 16 + lr;
#pragma unroll
            for (int r = 0; r < 8; ++r) yr[(size_t)r * N] = acc[i][t][r];
        }
}

// ---------------------------------------------------------------- split qkv -> q,k,v (B,H,T,hd) bf16
__global__ __launch_bounds__(256) void pack_qkv_kernel(const float* __restrict__ qkv,
                                                       bf16* __restrict__ q,
                                                       bf16* __restrict__ k,
                                                       bf16* __restrict__ v) {
    int i = blockIdx.x * blockDim.x + threadIdx.x;   // over B*T*3C
    if (i >= BATCH * SEQ * 3 * DMODEL) return;
    int f   = i % (3 * DMODEL);
    int row = i / (3 * DMODEL);
    int b = row / SEQ, t = row % SEQ;
    int h = f / (3 * HEADD);
    int j = f % (3 * HEADD);
    int which = j >> 6;          // 0=q 1=k 2=v
    int d = j & 63;
    bf16 val = (bf16)qkv[i];
    size_t dst = (((size_t)b * NHEADS + h) * SEQ + t) * HEADD + d;
    if (which == 0)      q[dst] = val;
    else if (which == 1) k[dst] = val;
    else                 v[dst] = val;
}

// ---------------------------------------------------------------- flash attention, one wave per (b,h,16-row q tile)
__global__ __launch_bounds__(32) void attn_kernel(const bf16* __restrict__ Q,
                                                  const bf16* __restrict__ K,
                                                  const bf16* __restrict__ V,
                                                  bf16* __restrict__ O) {
    __shared__ __align__(16) bf16 ldsP[16 * 32];

    int lane = threadIdx.x;
    int half = lane >> 4;
    int lr   = lane & 15;
    int qt = blockIdx.x;     // 0..127
    int h  = blockIdx.y;     // 0..15
    int b  = blockIdx.z;     // 0..1

    const bf16* qh = Q + ((size_t)b * NHEADS + h) * SEQ * HEADD;
    const bf16* kh = K + ((size_t)b * NHEADS + h) * SEQ * HEADD;
    const bf16* vh = V + ((size_t)b * NHEADS + h) * SEQ * HEADD;

    // Q tile A-fragments: row = lane&15, two hd-slices of 32
    Frag16 qa[2];
#pragma unroll
    for (int s = 0; s < 2; ++s) {
        const bf16* p = qh + (size_t)(qt * 16 + lr) * HEADD + s * 32 + half * 8;
        qa[s].q[0] = *(const uint4*)p;
        qa[s].q[1] = *(const uint4*)(p + 16);
    }

    v8f acc[4] = {{}, {}, {}, {}};           // out tile 16 x 64 (4 N-subtiles)
    float m[8], l[8];
#pragma unroll
    for (int r = 0; r < 8; ++r) { m[r] = -1.0e30f; l[r] = 0.0f; }

    int nk  = qt * 16 + 16;                  // causal: keys [0, nk)
    int nch = (nk + 31) >> 5;                // 32 keys per chunk

    for (int c = 0; c < nch; ++c) {
        int kbase = c << 5;

        // ---- QK^T : two 16x16 logit tiles (keys kbase..+15, kbase+16..+31)
        v8f s0 = {}, s1 = {};
#pragma unroll
        for (int s = 0; s < 2; ++s) {
            Frag16 kb0, kb1;
            const bf16* p0 = kh + (size_t)(kbase + lr) * HEADD + s * 32 + half * 8;
            const bf16* p1 = kh + (size_t)(kbase + 16 + lr) * HEADD + s * 32 + half * 8;
            kb0.q[0] = *(const uint4*)p0;  kb0.q[1] = *(const uint4*)(p0 + 16);
            kb1.q[0] = *(const uint4*)p1;  kb1.q[1] = *(const uint4*)(p1 + 16);
            s0 = __builtin_amdgcn_wmma_f32_16x16x32_bf16(false, qa[s].v, false, kb0.v,
                                                         (short)0, s0, false, false);
            s1 = __builtin_amdgcn_wmma_f32_16x16x32_bf16(false, qa[s].v, false, kb1.v,
                                                         (short)0, s1, false, false);
        }

        // ---- scale + causal mask + online softmax (rows r + half*8, cols lr / lr+16)
        float alpha[8];
#pragma unroll
        for (int r = 0; r < 8; ++r) {
            int qrow = qt * 16 + r + half * 8;
            float x0 = s0[r] * 0.125f;
            float x1 = s1[r] * 0.125f;
            if (kbase + lr      > qrow) x0 = -1.0e30f;
            if (kbase + 16 + lr > qrow) x1 = -1.0e30f;
            float cm = fmaxf(x0, x1);
#pragma unroll
            for (int off = 1; off < 16; off <<= 1)
                cm = fmaxf(cm, __shfl_xor(cm, off, 32));
            float mnew = fmaxf(m[r], cm);
            float a    = __expf(m[r] - mnew);
            float e0   = __expf(x0 - mnew);
            float e1   = __expf(x1 - mnew);
            float rs = e0 + e1;
#pragma unroll
            for (int off = 1; off < 16; off <<= 1)
                rs += __shfl_xor(rs, off, 32);
            l[r] = l[r] * a + rs;
            m[r] = mnew;
            alpha[r] = a;
            int row = r + half * 8;
            ldsP[row * 32 + lr]      = (bf16)e0;
            ldsP[row * 32 + 16 + lr] = (bf16)e1;
        }
#pragma unroll
        for (int t = 0; t < 4; ++t)
#pragma unroll
            for (int r = 0; r < 8; ++r) acc[t][r] *= alpha[r];

        __syncthreads();   // order LDS writes (C layout) before reads (A layout)

        // ---- P (A layout) from LDS: row = lane&15, K(key) per WMMA mapping
        Frag16 pa;
        const bf16* lp = &ldsP[lr * 32 + half * 8];
        pa.q[0] = *(const uint4*)lp;
        pa.q[1] = *(const uint4*)(lp + 16);

        // ---- PV : acc[t] += P(16x32) @ V(32x16) for each hd subtile t
#pragma unroll
        for (int t = 0; t < 4; ++t) {
            v16bf bv;
#pragma unroll
            for (int j = 0; j < 16; ++j) {
                int kl = half * 8 + (j < 8 ? j : j + 8);      // key within chunk
                bv[j] = vh[(size_t)(kbase + kl) * HEADD + t * 16 + lr];
            }
            acc[t] = __builtin_amdgcn_wmma_f32_16x16x32_bf16(false, pa.v, false, bv,
                                                             (short)0, acc[t], false, false);
        }
        __syncthreads();
    }

    // ---- normalize and store bf16 into (B,T,C) for the LM-head GEMM
    float inv[8];
#pragma unroll
    for (int r = 0; r < 8; ++r) inv[r] = 1.0f / l[r];
#pragma unroll
    for (int t = 0; t < 4; ++t)
#pragma unroll
        for (int r = 0; r < 8; ++r) {
            int qrow = qt * 16 + r + half * 8;
            int d = t * 16 + lr;
            O[((size_t)b * SEQ + qrow) * DMODEL + h * HEADD + d] = (bf16)(acc[t][r] * inv[r]);
        }
}

// ---------------------------------------------------------------- host launcher
extern "C" void kernel_launch(void* const* d_in, const int* in_sizes, int n_in,
                              void* d_out, int out_size, void* d_ws, size_t ws_size,
                              hipStream_t stream) {
    const int*   tok    = (const int*)d_in[0];
    const float* embed  = (const float*)d_in[1];
    const float* pos    = (const float*)d_in[2];
    const float* qkv_w  = (const float*)d_in[3];
    const float* lm_w   = (const float*)d_in[4];
    float*       logits = (float*)d_out;

    // workspace carve-up (all sizes multiples of 256 bytes)
    char* w = (char*)d_ws;
    bf16* xb    = (bf16*)w;  w += (size_t)BATCH * SEQ * DMODEL * 2;          // 8 MB
    bf16* qkvwb = (bf16*)w;  w += (size_t)3 * DMODEL * DMODEL * 2;           // 6 MB
    bf16* lmwb  = (bf16*)w;  w += (size_t)VOCAB * DMODEL * 2;                // 62.5 MB
    float* qkvf = (float*)w; w += (size_t)BATCH * SEQ * 3 * DMODEL * 4;      // 48 MB
    bf16* qb    = (bf16*)w;  w += (size_t)BATCH * SEQ * DMODEL * 2;          // 8 MB
    bf16* kb    = (bf16*)w;  w += (size_t)BATCH * SEQ * DMODEL * 2;          // 8 MB
    bf16* vb    = (bf16*)w;  w += (size_t)BATCH * SEQ * DMODEL * 2;          // 8 MB
    bf16* ob    = (bf16*)w;  w += (size_t)BATCH * SEQ * DMODEL * 2;          // 8 MB

    // 1. weight conversions to bf16 (async LDS staging path; n % 256 == 0)
    {
        int n = 3 * DMODEL * DMODEL;
        cvt_bf16_async_kernel<<<n / 256, 256, 0, stream>>>(qkv_w, qkvwb, n);
    }
    {
        int n = VOCAB * DMODEL;
        cvt_bf16_async_kernel<<<n / 256, 256, 0, stream>>>(lm_w, lmwb, n);
    }

    // 2. embedding gather + positional add -> bf16 activations
    {
        int n = BATCH * SEQ * DMODEL;
        embed_kernel<<<(n + 255) / 256, 256, 0, stream>>>(tok, embed, pos, xb);
    }

    // 3. QKV projection: (4096 x 1024) @ (3072 x 1024)^T -> fp32
    {
        int M = BATCH * SEQ, N = 3 * DMODEL, Kd = DMODEL;
        int tiles = (M >> 5) * (N >> 6);
        gemm_bf16_kernel<<<(tiles + 3) / 4, 128, 0, stream>>>(xb, qkvwb, qkvf, M, N, Kd);
    }

    // 4. split into q/k/v (B,H,T,hd) bf16
    {
        int n = BATCH * SEQ * 3 * DMODEL;
        pack_qkv_kernel<<<(n + 255) / 256, 256, 0, stream>>>(qkvf, qb, kb, vb);
    }

    // 5. causal flash attention
    {
        dim3 grid(SEQ / 16, NHEADS, BATCH);
        attn_kernel<<<grid, 32, 0, stream>>>(qb, kb, vb, ob);
    }

    // 6. LM head: (4096 x 1024) @ (32000 x 1024)^T -> fp32 logits
    {
        int M = BATCH * SEQ, N = VOCAB, Kd = DMODEL;
        int tiles = (M >> 5) * (N >> 6);
        gemm_bf16_kernel<<<(tiles + 3) / 4, 128, 0, stream>>>(ob, lmwb, logits, M, N, Kd);
    }
}
